// StressHead_15848429322594
// MI455X (gfx1250) — matrix operand order repaired and checked by reading, not compiled, not run on hardware
//
#include <hip/hip_runtime.h>
#include <hip/hip_bf16.h>

typedef __attribute__((ext_vector_type(16))) __bf16   v16bf;
typedef __attribute__((ext_vector_type(8)))  float    v8f;
typedef __attribute__((ext_vector_type(4)))  unsigned v4u;
typedef __attribute__((ext_vector_type(8)))  unsigned v8u;

// ---------------- workspace layout (bytes) ----------------
// 0      : acc[704] f32 : energy[64] | natoms[64] | stress[576]
// 4096   : colparams  (512 cols * 8 f32 = 16384 B) {b1,w2,m0,m1,m2,v0,v1,v2}
// 32768  : w1 packed bf16 B-fragments (512 KB)
// 557056 : forces float4[N]
#define WS_ACC      0
#define WS_COLP     4096
#define WS_W1PK     32768
#define WS_FORCES   557056

__device__ __forceinline__ unsigned bf16_rne(float x) {
  unsigned u = __float_as_uint(x);
  return (u + 0x7FFFu + ((u >> 16) & 1u)) >> 16;
}
__device__ __forceinline__ unsigned pack_bf16x2(float lo, float hi) {
  return bf16_rne(lo) | (bf16_rne(hi) << 16);
}

// ---- Tensor Data Mover: 1-D contiguous copy global -> LDS (nbytes % 8 == 0) ----
// D# per CDNA5 ISA: group0 {count=1, lds_addr, 57b global
// addr, type=2}; group1 {data_size=8B, tensor_dim0=tile_dim0=n8, dim1=1}.
__device__ __forceinline__ void tdm_load_1d(unsigned lds_addr, const void* gptr,
                                            unsigned n8 /* # of 8-byte elems */) {
  unsigned long long ga = (unsigned long long)(uintptr_t)gptr;
  v4u g0;
  g0[0] = 1u;                                             // count=1, user mode
  g0[1] = lds_addr;                                       // LDS byte address
  g0[2] = (unsigned)ga;                                   // global addr [31:0]
  g0[3] = (unsigned)((ga >> 32) & 0x1FFFFFFu) | (2u << 30); // addr[56:32] | type=2
  v8u g1;
  g1[0] = 3u << 16;                                       // data_size = 8 bytes
  g1[1] = (n8 & 0xFFFFu) << 16;                           // tensor_dim0 lo16
  g1[2] = (n8 >> 16) | (1u << 16);                        // tensor_dim0 hi | tensor_dim1=1
  g1[3] = (n8 & 0xFFFFu) << 16;                           // tile_dim0
  g1[4] = 1u;                                             // tile_dim1=1, tile_dim2=0
  g1[5] = n8;                                             // tensor_dim0_stride lo32
  g1[6] = 0u;
  g1[7] = 0u;
  asm volatile("tensor_load_to_lds %0, %1" :: "s"(g0), "s"(g1) : "memory");
}

// ---------------- kernel 0a: zero accumulators + column params ----------------
__global__ __launch_bounds__(256) void prep_kernel(
    const float* __restrict__ w_pos,   // [3,512]
    const float* __restrict__ w1,      // [512,512]
    const float* __restrict__ b1,      // [512]
    const float* __restrict__ w2,      // [512]
    float* __restrict__ acc,           // [704]
    float4* __restrict__ colp)         // [512*2] float4
{
  const int tid = threadIdx.x;
  for (int i = tid; i < 704; i += 256) acc[i] = 0.0f;
#pragma unroll
  for (int jj = 0; jj < 2; ++jj) {
    int j = tid * 2 + jj;              // 0..511
    float m0 = 0.f, m1 = 0.f, m2 = 0.f;
    for (int f = 0; f < 512; ++f) {
      float w = w1[f * 512 + j];
      m0 = fmaf(w_pos[f], w, m0);
      m1 = fmaf(w_pos[512 + f], w, m1);
      m2 = fmaf(w_pos[1024 + f], w, m2);
    }
    float w2j = w2[j];
    colp[j * 2 + 0] = make_float4(b1[j], w2j, m0, m1);
    colp[j * 2 + 1] = make_float4(m2, m0 * w2j, m1 * w2j, m2 * w2j);
  }
}

// ---------------- kernel 0b: repack w1 (f32) -> bf16 B-fragment layout -------
// dword index = ((nt*16 + kt)*32 + lane)*8 + r
// lane: n = nt*16 + (lane&15), khalf = lane>>4 ; dword r holds K = kt*32+khalf*16+2r (lo), +1 (hi)
__global__ __launch_bounds__(256) void pack_w1_kernel(
    const float* __restrict__ w1, unsigned* __restrict__ w1pk)
{
  int idx = blockIdx.x * 256 + threadIdx.x;     // 0 .. 131071
  int r    =  idx        & 7;
  int lane = (idx >> 3)  & 31;
  int kt   = (idx >> 8)  & 15;
  int nt   =  idx >> 12;
  int n  = nt * 16 + (lane & 15);
  int K0 = kt * 32 + (lane >> 4) * 16 + 2 * r;
  w1pk[idx] = pack_bf16x2(w1[K0 * 512 + n], w1[(K0 + 1) * 512 + n]);
}

// ---------------- kernel 1: fused GEMM (bf16 WMMA) + silu/silu' epilogue -----
__global__ __launch_bounds__(256) void node_kernel(
    const float* __restrict__ node_feats,  // [N,512]
    const float* __restrict__ pos,         // [N,3]
    const int*   __restrict__ batch,       // [N]
    const float* __restrict__ b2p,         // [1]
    const float4* __restrict__ colp,       // [512*2]
    const int4*  __restrict__ w1pk,        // packed bf16 fragments
    float4* __restrict__ forces,           // [N]
    float* __restrict__ energy_acc,        // [64]
    float* __restrict__ natoms_acc,        // [64]
    int N)
{
  __shared__ v16bf Ash[8 * 16 * 32];   // 128 KB: [tile][kt][lane] A-fragments (bf16)
  __shared__ v16bf Bsh[2][16 * 32];    // 2 x 16 KB: double-buffered B slice

  const int tid  = threadIdx.x;
  const int lane = tid & 31;
  const int wave = tid >> 5;
  const int half = lane >> 4;
  const int rowbase  = blockIdx.x * 128;
  const int tilebase = rowbase + wave * 16;

  // kick off TDM for the first B slice while we stage A
  if (wave == 0)
    tdm_load_1d((unsigned)(uintptr_t)&Bsh[0][0], w1pk, 2048);

  // ---- stage A: rows [rowbase, rowbase+128) of node_feats -> bf16 fragments in LDS
  {
    int* AshW = reinterpret_cast<int*>(Ash);
    for (int it = 0; it < 64; ++it) {
      int linear = it * 256 + tid;          // 0..16383 (128 float4 per row)
      int row_local = linear >> 7;
      int q = linear & 127;                 // float4 index within row
      int grow = rowbase + row_local;
      float4 v = make_float4(0.f, 0.f, 0.f, 0.f);
      if (grow < N) v = *(const float4*)(node_feats + (size_t)grow * 512 + 4 * q);
      int tile = row_local >> 4, m = row_local & 15;
      int kp = 2 * q;                       // even K-pair index
      int kt = kp >> 4, c = kp & 15;        // c even -> r even, r+1 same fragment
      int h  = c >> 3;
      int r  = c & 7;
      int idx = (((tile * 16 + kt) * 32) + h * 16 + m) * 8 + r;
      uint2 d;
      d.x = pack_bf16x2(v.x, v.y);
      d.y = pack_bf16x2(v.z, v.w);
      *(uint2*)(AshW + idx) = d;            // 8-byte aligned (r even)
    }
  }

  // ---- per-lane row data (C/D layout: VGPR r <-> row half*8 + r)
  float px[8], py[8], pz[8];
#pragma unroll
  for (int r = 0; r < 8; ++r) {
    int gr = tilebase + half * 8 + r;
    if (gr < N) { px[r] = pos[3*gr]; py[r] = pos[3*gr+1]; pz[r] = pos[3*gr+2]; }
    else        { px[r] = 0.f; py[r] = 0.f; pz[r] = 0.f; }
  }
  float acc_e[8], f0[8], f1[8], f2[8];
#pragma unroll
  for (int r = 0; r < 8; ++r) { acc_e[r]=0.f; f0[r]=0.f; f1[r]=0.f; f2[r]=0.f; }

  if (wave == 0) __builtin_amdgcn_s_wait_tensorcnt(0);   // B slice 0 landed
  __syncthreads();

#pragma unroll 1
  for (int nt = 0; nt < 32; ++nt) {
    const int cur = nt & 1;
    // prefetch next B slice into the other buffer (overlaps with compute below)
    if (wave == 0 && nt + 1 < 32)
      tdm_load_1d((unsigned)(uintptr_t)&Bsh[cur ^ 1][0], w1pk + (nt + 1) * 1024, 2048);

    int ncol = nt * 16 + (lane & 15);
    float4 cpa = colp[ncol * 2 + 0];       // {b1, w2, m0, m1}
    float4 cpb = colp[ncol * 2 + 1];       // {m2, v0, v1, v2}

    // two independent accumulator chains -> 2 WMMAs in flight, deeper ds pipelining
    v8f C0 = {0.f,0.f,0.f,0.f,0.f,0.f,0.f,0.f};
    v8f C1 = {0.f,0.f,0.f,0.f,0.f,0.f,0.f,0.f};
#pragma unroll
    for (int kt = 0; kt < 16; kt += 2) {
      v16bf a0 = Ash[(wave * 16 + kt) * 32 + lane];
      v16bf b0 = Bsh[cur][kt * 32 + lane];
      C0 = __builtin_amdgcn_wmma_f32_16x16x32_bf16(false, a0, false, b0,
                                                   (short)0, C0, false, false);
      v16bf a1 = Ash[(wave * 16 + kt + 1) * 32 + lane];
      v16bf b1 = Bsh[cur][(kt + 1) * 32 + lane];
      C1 = __builtin_amdgcn_wmma_f32_16x16x32_bf16(false, a1, false, b1,
                                                   (short)0, C1, false, false);
    }
    // epilogue: z -> silu (energy), silu' (forces)
#pragma unroll
    for (int r = 0; r < 8; ++r) {
      float z = (C0[r] + C1[r]) + cpa.x;
      z = fmaf(px[r], cpa.z, z);
      z = fmaf(py[r], cpa.w, z);
      z = fmaf(pz[r], cpb.x, z);
      float s = __builtin_amdgcn_rcpf(1.0f + __expf(-z));   // sigmoid
      acc_e[r] = fmaf(cpa.y, z * s, acc_e[r]);              // + w2 * silu(z)
      float g = fmaf(z * s, 1.0f - s, s);                   // silu'(z)
      f0[r] = fmaf(cpb.y, g, f0[r]);
      f1[r] = fmaf(cpb.z, g, f1[r]);
      f2[r] = fmaf(cpb.w, g, f2[r]);
    }

    if (wave == 0) __builtin_amdgcn_s_wait_tensorcnt(0);   // next slice landed
    __syncthreads();                                       // release cur, publish next
  }

  // ---- butterfly reduce over the 16 lanes sharing each row half
#pragma unroll
  for (int m = 1; m <= 8; m <<= 1) {
#pragma unroll
    for (int r = 0; r < 8; ++r) {
      acc_e[r] += __shfl_xor(acc_e[r], m, 32);
      f0[r]    += __shfl_xor(f0[r],    m, 32);
      f1[r]    += __shfl_xor(f1[r],    m, 32);
      f2[r]    += __shfl_xor(f2[r],    m, 32);
    }
  }

  float b2v = b2p[0];
#pragma unroll
  for (int r = 0; r < 8; ++r) {
    if ((lane & 15) == 0) {                // lanes 0 and 16 hold the full sums
      int gr = tilebase + half * 8 + r;
      if (gr < N) {
        forces[gr] = make_float4(f0[r], f1[r], f2[r], 0.f);
        int g = batch[gr];
        atomicAdd(&energy_acc[g], acc_e[r] + b2v);
        atomicAdd(&natoms_acc[g], 1.0f);
      }
    }
  }
}

// ---------------- kernel 2: edge virial scatter with LDS stress bins ---------
#define EDGES_PER_BLOCK 4096
__global__ __launch_bounds__(256) void edge_kernel(
    const int*   __restrict__ ei,        // [2,E]
    const float* __restrict__ edge_attr, // [E,3]
    const int*   __restrict__ batch,     // [N]
    const float4* __restrict__ forces,   // [N]
    float* __restrict__ stress_acc,      // [576]
    int E)
{
  __shared__ float s[2][576];            // 2 copies to halve ds-atomic contention
  for (int i = threadIdx.x; i < 1152; i += 256) s[0][i] = 0.f;
  __syncthreads();

  const int cpy = (threadIdx.x >> 5) & 1;
  int base = blockIdx.x * EDGES_PER_BLOCK;
#pragma unroll 1
  for (int it = 0; it < EDGES_PER_BLOCK / 256; ++it) {
    int e = base + it * 256 + threadIdx.x;
    if (e < E) {
      int row = ei[e];
      int col = ei[(size_t)E + e];
      int g = batch[row];
      float4 f = forces[col];
      float a0 = edge_attr[(size_t)e * 3 + 0];
      float a1 = edge_attr[(size_t)e * 3 + 1];
      float a2 = edge_attr[(size_t)e * 3 + 2];
      float* sg = s[cpy] + g * 9;
      atomicAdd(sg + 0, a0 * f.x); atomicAdd(sg + 1, a0 * f.y); atomicAdd(sg + 2, a0 * f.z);
      atomicAdd(sg + 3, a1 * f.x); atomicAdd(sg + 4, a1 * f.y); atomicAdd(sg + 5, a1 * f.z);
      atomicAdd(sg + 6, a2 * f.x); atomicAdd(sg + 7, a2 * f.y); atomicAdd(sg + 8, a2 * f.z);
    }
  }
  __syncthreads();
  for (int i = threadIdx.x; i < 576; i += 256)
    atomicAdd(&stress_acc[i], s[0][i] + s[1][i]);
}

// ---------------- kernel 3: finalize -----------------------------------------
__global__ void finalize_kernel(
    const float* __restrict__ cell,   // [64,3,3]
    const float* __restrict__ acc,    // energy[64] | natoms[64] | stress[576]
    float* __restrict__ out, int G)
{
  int g = threadIdx.x;
  if (g < G) {
    const float* c = cell + g * 9;
    float det = c[0]*(c[4]*c[8]-c[5]*c[7])
              - c[1]*(c[3]*c[8]-c[5]*c[6])
              + c[2]*(c[3]*c[7]-c[4]*c[6]);
    float vol = fmaxf(fabsf(det), 1e-10f);
    out[g]       = acc[g];
    out[G + g]   = acc[64 + g];
#pragma unroll
    for (int k = 0; k < 9; ++k)
      out[2 * G + g * 9 + k] = acc[128 + g * 9 + k] / vol;
  }
}

// ---------------- launch ------------------------------------------------------
extern "C" void kernel_launch(void* const* d_in, const int* in_sizes, int n_in,
                              void* d_out, int out_size, void* d_ws, size_t ws_size,
                              hipStream_t stream) {
  const float* node_feats = (const float*)d_in[0];
  const float* pos        = (const float*)d_in[1];
  const float* edge_attr  = (const float*)d_in[2];
  const float* cell       = (const float*)d_in[3];
  const int*   batch      = (const int*)d_in[4];
  const int*   edge_index = (const int*)d_in[5];
  const float* w_pos      = (const float*)d_in[6];
  const float* w1         = (const float*)d_in[7];
  const float* b1         = (const float*)d_in[8];
  const float* w2         = (const float*)d_in[9];
  const float* b2         = (const float*)d_in[10];

  const int N = in_sizes[0] / 512;
  const int E = in_sizes[2] / 3;
  const int G = in_sizes[3] / 9;

  char* ws = (char*)d_ws;
  float*    acc    = (float*)(ws + WS_ACC);
  float4*   colp   = (float4*)(ws + WS_COLP);
  unsigned* w1pk   = (unsigned*)(ws + WS_W1PK);
  float4*   forces = (float4*)(ws + WS_FORCES);

  prep_kernel<<<1, 256, 0, stream>>>(w_pos, w1, b1, w2, acc, colp);
  pack_w1_kernel<<<512, 256, 0, stream>>>(w1, w1pk);

  int node_blocks = (N + 127) / 128;
  node_kernel<<<node_blocks, 256, 0, stream>>>(
      node_feats, pos, batch, b2, colp, (const int4*)w1pk,
      forces, acc, acc + 64, N);

  int edge_blocks = (E + EDGES_PER_BLOCK - 1) / EDGES_PER_BLOCK;
  edge_kernel<<<edge_blocks, 256, 0, stream>>>(
      edge_index, edge_attr, batch, forces, acc + 128, E);

  finalize_kernel<<<1, 64, 0, stream>>>(cell, acc, (float*)d_out, G);
}
